// DSCLayer_9405978378747
// MI455X (gfx1250) — matrix operand order repaired and checked by reading, not compiled, not run on hardware
//
#include <hip/hip_runtime.h>
#include <hip/hip_bf16.h>
#include <math.h>

// ---------------------------------------------------------------------------
// Types for CDNA5 WMMA
// ---------------------------------------------------------------------------
typedef __attribute__((ext_vector_type(16))) _Float16 v16h;
typedef __attribute__((ext_vector_type(8)))  float    v8f;

#define TAU 10.0f
#define EPSV 1e-6f
#define TOPK 8

// GEMM tiling: 128x128 block tile, K-step 32, 256 threads = 8 waves (wave32).
// Wave grid 2(M) x 4(N): each wave owns 64x32 = 4x2 WMMA 16x16 tiles.
#define BM 128
#define BN 128
#define BK 32
#define LDSS 40   // LDS row stride in halves (80B, multiple of 16B)

// ---------------------------------------------------------------------------
// f32 -> f16 conversion
// ---------------------------------------------------------------------------
__global__ __launch_bounds__(256)
void f32_to_f16_kernel(const float* __restrict__ in, _Float16* __restrict__ out,
                       long long n) {
  long long i = (long long)blockIdx.x * 256 + threadIdx.x;
  if (i < n) out[i] = (_Float16)in[i];
}

// ---------------------------------------------------------------------------
// Per-token: LayerNorm(x) and x, both to f16 (one read of x)
// ---------------------------------------------------------------------------
__global__ __launch_bounds__(256)
void prep_x_kernel(const float* __restrict__ x, const float* __restrict__ g,
                   const float* __restrict__ b, _Float16* __restrict__ Xh,
                   _Float16* __restrict__ XLNh, int D) {
  __shared__ float rs[256];
  __shared__ float rs2[256];
  const int tid = threadIdx.x;
  const long long n = blockIdx.x;
  const float* xr = x + n * D;
  float s = 0.f, s2 = 0.f;
  for (int d = tid; d < D; d += 256) { float v = xr[d]; s += v; s2 += v * v; }
  rs[tid] = s; rs2[tid] = s2;
  for (int off = 128; off > 0; off >>= 1) {
    __syncthreads();
    if (tid < off) { rs[tid] += rs[tid + off]; rs2[tid] += rs2[tid + off]; }
  }
  __syncthreads();
  const float mean = rs[0] / (float)D;
  const float var  = rs2[0] / (float)D - mean * mean;
  const float rstd = rsqrtf(var + 1e-5f);
  for (int d = tid; d < D; d += 256) {
    float v = xr[d];
    Xh[n * D + d]   = (_Float16)v;
    XLNh[n * D + d] = (_Float16)((v - mean) * rstd * g[d] + b[d]);
  }
}

// ---------------------------------------------------------------------------
// Unit-normalize rows of a [R, D] matrix
// ---------------------------------------------------------------------------
__global__ __launch_bounds__(256)
void unit_rows_kernel(const float* __restrict__ raw, float* __restrict__ out,
                      int D) {
  __shared__ float rs[256];
  const int tid = threadIdx.x;
  const long long r = blockIdx.x;
  const float* row = raw + r * D;
  float s = 0.f;
  for (int d = tid; d < D; d += 256) { float v = row[d]; s += v * v; }
  rs[tid] = s;
  for (int off = 128; off > 0; off >>= 1) {
    __syncthreads();
    if (tid < off) rs[tid] += rs[tid + off];
  }
  __syncthreads();
  const float inv = 1.f / fmaxf(sqrtf(rs[0]), EPSV);
  for (int d = tid; d < D; d += 256) out[r * D + d] = row[d] * inv;
}

// ---------------------------------------------------------------------------
// WMMA NT-GEMM: C[M,N] = A[M,K] * B[N,K]^T  (both row-major over K)
// Double-buffered LDS (ping-pong), one barrier per K-step, all fragments
// hoisted so the 8 WMMAs issue back-to-back into the matrix pipe.
// EPI 0: router  -> fout = softplus(clip(acc + extra[col], +/-TAU))
// EPI 1: gelu    -> hout = (f16) gelu(acc)
// EPI 2: add dyn -> fout = acc + extra[row*N+col]   (extra aliases fout)
// Requires M%128==0, N%128==0, K%32==0.
// ---------------------------------------------------------------------------
template <int EPI>
__global__ __launch_bounds__(256)
void wmma_gemm_nt(const _Float16* __restrict__ A, const _Float16* __restrict__ B,
                  int K, int N, float* fout, _Float16* hout, const float* extra) {
  __shared__ __align__(16) _Float16 As[2][BM * LDSS];
  __shared__ __align__(16) _Float16 Bs[2][BN * LDSS];

  const int tid    = threadIdx.x;
  const int lane   = tid & 31;
  const int wave   = tid >> 5;
  const int l16    = lane & 15;
  const int laneHi = lane >> 4;          // 0 or 1
  const int rowBase = (wave >> 2) * 64;  // wave M origin in tile
  const int colBase = (wave & 3) * 32;   // wave N origin in tile

  const long long blockRow = (long long)blockIdx.y * BM;
  const long long blockCol = (long long)blockIdx.x * BN;

  // Each thread copies 2 16B chunks of A and 2 of B per K-tile.
  const int ch0 = tid * 2, ch1 = tid * 2 + 1;
  const int rA0 = ch0 >> 2, cA0 = (ch0 & 3) * 8;
  const int rA1 = ch1 >> 2, cA1 = (ch1 & 3) * 8;
  const _Float16* gA0 = A + (blockRow + rA0) * (long long)K + cA0;
  const _Float16* gA1 = A + (blockRow + rA1) * (long long)K + cA1;
  const _Float16* gB0 = B + (blockCol + rA0) * (long long)K + cA0;
  const _Float16* gB1 = B + (blockCol + rA1) * (long long)K + cA1;

  v8f acc[4][2] = {};

  auto compute = [&](int p) {
    // B fragments (32x16 KxN): lane holds column N=l16; K = laneHi*16 .. +16.
    v16h bf[2];
#pragma unroll
    for (int j = 0; j < 2; ++j) {
      const _Float16* bp = &Bs[p][(colBase + j * 16 + l16) * LDSS + laneHi * 16];
      ((uint4*)&bf[j])[0] = *(const uint4*)(bp);
      ((uint4*)&bf[j])[1] = *(const uint4*)(bp + 8);
    }
    // A fragments (16x32 MxK): lane holds row M=l16;
    // elems 0..15 -> K = laneHi*16 .. +16
    v16h af[4];
#pragma unroll
    for (int i = 0; i < 4; ++i) {
      const _Float16* ap = &As[p][(rowBase + i * 16 + l16) * LDSS];
      ((uint4*)&af[i])[0] = *(const uint4*)(ap + laneHi * 16);
      ((uint4*)&af[i])[1] = *(const uint4*)(ap + laneHi * 16 + 8);
    }
    // 8 back-to-back WMMAs (distinct source/dest regs -> dense XDL issue).
#pragma unroll
    for (int i = 0; i < 4; ++i)
#pragma unroll
      for (int j = 0; j < 2; ++j)
        acc[i][j] = __builtin_amdgcn_wmma_f32_16x16x32_f16(
            false, af[i], false, bf[j], (short)0, acc[i][j], false, false);
  };

  const int numK = K / BK;
  // Preload K-tile 0 into buffer 0.
  {
    const uint4 a0 = *(const uint4*)(gA0);
    const uint4 a1 = *(const uint4*)(gA1);
    const uint4 b0 = *(const uint4*)(gB0);
    const uint4 b1 = *(const uint4*)(gB1);
    *(uint4*)(&As[0][rA0 * LDSS + cA0]) = a0;
    *(uint4*)(&As[0][rA1 * LDSS + cA1]) = a1;
    *(uint4*)(&Bs[0][rA0 * LDSS + cA0]) = b0;
    *(uint4*)(&Bs[0][rA1 * LDSS + cA1]) = b1;
  }
  __syncthreads();

  for (int kt = 0; kt < numK - 1; ++kt) {
    const int p = kt & 1;
    const long long k1 = (long long)(kt + 1) * BK;
    // Issue next tile's global loads before compute (overlap with WMMA).
    const uint4 a0 = *(const uint4*)(gA0 + k1);
    const uint4 a1 = *(const uint4*)(gA1 + k1);
    const uint4 b0 = *(const uint4*)(gB0 + k1);
    const uint4 b1 = *(const uint4*)(gB1 + k1);
    // Speculative prefetch of the K+2 tile (global_prefetch_b8; OOB is dropped).
    __builtin_prefetch(gA0 + k1 + BK, 0, 1);
    __builtin_prefetch(gB0 + k1 + BK, 0, 1);

    compute(p);

    // Stage next tile into the other buffer.
    *(uint4*)(&As[1 - p][rA0 * LDSS + cA0]) = a0;
    *(uint4*)(&As[1 - p][rA1 * LDSS + cA1]) = a1;
    *(uint4*)(&Bs[1 - p][rA0 * LDSS + cA0]) = b0;
    *(uint4*)(&Bs[1 - p][rA1 * LDSS + cA1]) = b1;
    __syncthreads();
  }
  compute((numK - 1) & 1);

  // Epilogue. C layout: VGPR r holds row r + laneHi*8, col = l16.
#pragma unroll
  for (int i = 0; i < 4; ++i) {
#pragma unroll
    for (int j = 0; j < 2; ++j) {
#pragma unroll
      for (int r = 0; r < 8; ++r) {
        const long long row = blockRow + rowBase + i * 16 + laneHi * 8 + r;
        const long long col = blockCol + colBase + j * 16 + l16;
        const float v = acc[i][j][r];
        if (EPI == 0) {
          float rr = fminf(fmaxf(v + extra[col], -TAU), TAU);
          fout[row * N + col] = fmaxf(rr, 0.f) + log1pf(expf(-fabsf(rr)));
        } else if (EPI == 1) {
          const float ge = 0.5f * v * (1.f + erff(v * 0.70710678118654752f));
          hout[row * N + col] = (_Float16)ge;
        } else {
          fout[row * N + col] = v + extra[row * N + col];
        }
      }
    }
  }
}

// ---------------------------------------------------------------------------
// Top-8 per token over NB router outputs; one wave32 per token.
// ---------------------------------------------------------------------------
__global__ __launch_bounds__(256)
void topk_kernel(const float* __restrict__ alpha, int* __restrict__ idxOut,
                 float* __restrict__ zOut, int NB) {
  const int wave = threadIdx.x >> 5;
  const int lane = threadIdx.x & 31;
  const long long n = (long long)blockIdx.x * 8 + wave;
  const float* a = alpha + n * NB;
  const int per = NB / 32;  // 16 for NB=512
  float vals[16];
  for (int i = 0; i < per; ++i) vals[i] = a[lane + 32 * i];

  float phi[TOPK];
  int   pid[TOPK];
  float ssum = 0.f;
  for (int k = 0; k < TOPK; ++k) {
    float mv = -1e30f; int mi = 0x7fffffff;
    for (int i = 0; i < per; ++i)
      if (vals[i] > mv) { mv = vals[i]; mi = lane + 32 * i; }
    for (int off = 16; off > 0; off >>= 1) {
      const float ov = __shfl_xor(mv, off, 32);
      const int   oi = __shfl_xor(mi, off, 32);
      if (ov > mv || (ov == mv && oi < mi)) { mv = ov; mi = oi; }
    }
    phi[k] = mv; pid[k] = mi; ssum += mv;
    if ((mi & 31) == lane) vals[mi >> 5] = -1e30f;  // mask winner
  }
  if (lane == 0) {
    const float scale = tanhf(ssum) / (ssum + EPSV);
    for (int k = 0; k < TOPK; ++k) {
      idxOut[n * TOPK + k] = pid[k];
      zOut[n * TOPK + k]   = phi[k] * scale;
    }
  }
}

// ---------------------------------------------------------------------------
// Dynamic path: block per token. h_lat_k = x . Uunit[idx_k];
// dyn[d] = gamma[d] * sum_k (h_lat_k * Z_k) * Vunit[idx_k][d]  -> d_out
// ---------------------------------------------------------------------------
__global__ __launch_bounds__(256)
void dyn_kernel(const float* __restrict__ x, const int* __restrict__ idx,
                const float* __restrict__ Z, const float* __restrict__ Uu,
                const float* __restrict__ Vu, const float* __restrict__ gamma,
                float* __restrict__ out, int D) {
  __shared__ float xs[1024];
  __shared__ float red[256];
  __shared__ float coef[TOPK];
  const int tid = threadIdx.x;
  const long long n = blockIdx.x;
  const float* xr = x + n * D;
  for (int d = tid; d < D; d += 256) xs[d] = xr[d];
  __syncthreads();
  for (int k = 0; k < TOPK; ++k) {
    const long long id = idx[n * TOPK + k];
    const float* u = Uu + id * D;
    float s = 0.f;
    for (int d = tid; d < D; d += 256) s += xs[d] * u[d];
    red[tid] = s;
    for (int off = 128; off > 0; off >>= 1) {
      __syncthreads();
      if (tid < off) red[tid] += red[tid + off];
    }
    __syncthreads();
    if (tid == 0) coef[k] = red[0] * Z[n * TOPK + k];
    __syncthreads();
  }
  for (int d = tid; d < D; d += 256) {
    float acc = 0.f;
#pragma unroll
    for (int k = 0; k < TOPK; ++k) {
      const long long id = idx[n * TOPK + k];
      acc += coef[k] * Vu[id * D + d];
    }
    out[n * D + d] = acc * gamma[d];
  }
}

// ---------------------------------------------------------------------------
// Launch
// ---------------------------------------------------------------------------
extern "C" void kernel_launch(void* const* d_in, const int* in_sizes, int n_in,
                              void* d_out, int out_size, void* d_ws, size_t ws_size,
                              hipStream_t stream) {
  const float* x     = (const float*)d_in[0];
  const float* W1    = (const float*)d_in[1];
  const float* W2    = (const float*)d_in[2];
  const float* ln_g  = (const float*)d_in[3];
  const float* ln_b  = (const float*)d_in[4];
  const float* rW    = (const float*)d_in[5];
  const float* rb    = (const float*)d_in[6];
  const float* rawU  = (const float*)d_in[7];
  const float* rawV  = (const float*)d_in[8];
  const float* gamma = (const float*)d_in[9];

  const int D  = in_sizes[3];                 // 1024
  const int H  = in_sizes[1] / D;             // 4096
  const int NB = in_sizes[5] / D;             // 512
  const long long Ntok = (long long)in_sizes[0] / D;  // 8192
  float* out = (float*)d_out;

  char* ws = (char*)d_ws;
  size_t off = 0;
  auto alloc = [&](size_t bytes) -> void* {
    void* p = ws + off;
    off += (bytes + 255) & ~(size_t)255;
    return p;
  };
  _Float16* Xh   = (_Float16*)alloc((size_t)Ntok * D * 2);
  _Float16* XLNh = (_Float16*)alloc((size_t)Ntok * D * 2);
  _Float16* W1h  = (_Float16*)alloc((size_t)H * D * 2);
  _Float16* W2h  = (_Float16*)alloc((size_t)D * H * 2);
  _Float16* RWh  = (_Float16*)alloc((size_t)NB * D * 2);
  _Float16* Gh   = (_Float16*)alloc((size_t)Ntok * H * 2);
  float*    alpha= (float*)   alloc((size_t)Ntok * NB * 4);
  float*    Uu   = (float*)   alloc((size_t)NB * D * 4);
  float*    Vu   = (float*)   alloc((size_t)NB * D * 4);
  int*      idx  = (int*)     alloc((size_t)Ntok * TOPK * 4);
  float*    Z    = (float*)   alloc((size_t)Ntok * TOPK * 4);
  (void)ws_size; (void)n_in; (void)out_size;

  // Precision conversions
  const long long nW1 = (long long)H * D, nW2 = (long long)D * H, nRW = (long long)NB * D;
  f32_to_f16_kernel<<<(int)((nW1 + 255) / 256), 256, 0, stream>>>(W1, W1h, nW1);
  f32_to_f16_kernel<<<(int)((nW2 + 255) / 256), 256, 0, stream>>>(W2, W2h, nW2);
  f32_to_f16_kernel<<<(int)((nRW + 255) / 256), 256, 0, stream>>>(rW, RWh, nRW);

  // LayerNorm + x -> f16
  prep_x_kernel<<<(int)Ntok, 256, 0, stream>>>(x, ln_g, ln_b, Xh, XLNh, D);

  // Unit rows of U, V
  unit_rows_kernel<<<NB, 256, 0, stream>>>(rawU, Uu, D);
  unit_rows_kernel<<<NB, 256, 0, stream>>>(rawV, Vu, D);

  // Router GEMM + clip + softplus (WMMA)
  wmma_gemm_nt<0><<<dim3(NB / BN, (int)(Ntok / BM)), 256, 0, stream>>>(
      XLNh, RWh, D, NB, alpha, nullptr, rb);

  // Top-8 + Z
  topk_kernel<<<(int)(Ntok / 8), 256, 0, stream>>>(alpha, idx, Z, NB);

  // Dynamic path -> d_out
  dyn_kernel<<<(int)Ntok, 256, 0, stream>>>(x, idx, Z, Uu, Vu, gamma, out, D);

  // Static path GEMM1: gelu(X @ W1^T) -> Gh (f16), WMMA
  wmma_gemm_nt<1><<<dim3(H / BN, (int)(Ntok / BM)), 256, 0, stream>>>(
      Xh, W1h, D, H, nullptr, Gh, nullptr);

  // Static path GEMM2: Gh @ W2^T + dyn -> d_out, WMMA
  wmma_gemm_nt<2><<<dim3(D / BN, (int)(Ntok / BM)), 256, 0, stream>>>(
      Gh, W2h, H, D, out, nullptr, out);
}
